// RegressorHybrid_46437186404761
// MI455X (gfx1250) — matrix operand order repaired and checked by reading, not compiled, not run on hardware
//
#include <hip/hip_runtime.h>

typedef __attribute__((ext_vector_type(16))) _Float16 v16h;
typedef __attribute__((ext_vector_type(8)))  _Float16 v8h;
typedef __attribute__((ext_vector_type(8)))  float    v8f;

#define WAVES_PER_BLOCK 2
#define THREADS (WAVES_PER_BLOCK * 32)

// LDS weight layout (half-element offsets), per-MLP stride
#define OW1 0        // 64 x 128
#define OW2 8192     // 64 x 64
#define OW3 12288    // 32 x 64
#define OW4 14336    // 16 x 32 (padded: only row n==0 nonzero)
#define WSTRIDE 14848
#define BSTRIDE 161  // floats: b1[64] b2[64] b3[32] b4[1]

__device__ __forceinline__ float lrelu(float x) { return x > 0.f ? x : 0.01f * x; }

__device__ __forceinline__ v8f wmma16(v16h a, v16h b, v8f c) {
  return __builtin_amdgcn_wmma_f32_16x16x32_f16(false, a, false, b, (short)0, c, false, false);
}

__global__ __launch_bounds__(THREADS) void linkmlp_kernel(
    const float* __restrict__ x_src, const float* __restrict__ x_dst,
    const int* __restrict__ eli,
    const float* __restrict__ ew1, const float* __restrict__ eb1,
    const float* __restrict__ ww1, const float* __restrict__ wb1,
    const float* __restrict__ ew2, const float* __restrict__ eb2,
    const float* __restrict__ ww2, const float* __restrict__ wb2,
    const float* __restrict__ ew3, const float* __restrict__ eb3,
    const float* __restrict__ ww3, const float* __restrict__ wb3,
    const float* __restrict__ ew4, const float* __restrict__ eb4,
    const float* __restrict__ ww4, const float* __restrict__ wb4,
    float* __restrict__ out, int E)
{
  __shared__ __align__(64) _Float16 sW[2 * WSTRIDE];
  __shared__ __align__(64) float    sB[2 * BSTRIDE];
  __shared__ __align__(64) _Float16 sAct[WAVES_PER_BLOCK][16 * 64];

  const int tid = threadIdx.x;

  // ---- stage weights into LDS, transposed to [n][k] f16 ----
  auto stage = [&](int dstOff, const float* w, int din, int dout) {
    for (int i = tid; i < din * dout; i += THREADS) {
      int n = i / din;
      int k = i - n * din;
      sW[dstOff + i] = (_Float16)w[(size_t)k * dout + n];
    }
  };
  stage(OW1, ew1, 128, 64);
  stage(OW2, ew2, 64, 64);
  stage(OW3, ew3, 64, 32);
  stage(WSTRIDE + OW1, ww1, 128, 64);
  stage(WSTRIDE + OW2, ww2, 64, 64);
  stage(WSTRIDE + OW3, ww3, 64, 32);
  for (int i = tid; i < 512; i += THREADS) {          // layer-4 padded 16x32
    int n = i >> 5, k = i & 31;
    sW[OW4 + i]           = (n == 0) ? (_Float16)ew4[k] : (_Float16)0.f;
    sW[WSTRIDE + OW4 + i] = (n == 0) ? (_Float16)ww4[k] : (_Float16)0.f;
  }
  for (int i = tid; i < 64; i += THREADS) {
    sB[i]      = eb1[i];  sB[64 + i]           = eb2[i];
    sB[BSTRIDE + i] = wb1[i];  sB[BSTRIDE + 64 + i] = wb2[i];
  }
  for (int i = tid; i < 32; i += THREADS) {
    sB[128 + i] = eb3[i];  sB[BSTRIDE + 128 + i] = wb3[i];
  }
  if (tid == 0) { sB[160] = eb4[0]; sB[BSTRIDE + 160] = wb4[0]; }
  __syncthreads();

  const int lane = tid & 31;
  const int half = lane >> 4;   // which half-wave
  const int lm   = lane & 15;   // M row (A/output) or N col (B/C)
  const int wv   = tid >> 5;
  _Float16* act  = sAct[wv];

  const int numTiles  = (E + 15) >> 4;
  const int waveId    = blockIdx.x * WAVES_PER_BLOCK + wv;
  const int waveCount = gridDim.x * WAVES_PER_BLOCK;

  for (int tile = waveId; tile < numTiles; tile += waveCount) {
    const int tileBase = tile << 4;
    if (tile + waveCount < numTiles) {
      __builtin_prefetch(eli + ((tile + waveCount) << 4), 0, 1);
      __builtin_prefetch(eli + E + ((tile + waveCount) << 4), 0, 1);
    }
    int e = tileBase + lm;
    if (e >= E) e = E - 1;
    const int sIdx = eli[e];
    const int dIdx = eli[E + e];
    const float* rowS = x_src + (size_t)sIdx * 64;
    const float* rowD = x_dst + (size_t)dIdx * 64;

    // ---- build layer-1 A operands (16 edges x 128 feats, f32 -> f16) ----
    v16h A1[4];
#pragma unroll
    for (int kt = 0; kt < 4; ++kt) {
      const float* row = (kt < 2) ? rowS : rowD;
#pragma unroll
      for (int c = 0; c < 2; ++c) {
        const float4* p = (const float4*)(row + (kt & 1) * 32 + half * 16 + c * 8);
        float4 u = p[0], v = p[1];
        A1[kt][c * 8 + 0] = (_Float16)u.x;  A1[kt][c * 8 + 1] = (_Float16)u.y;
        A1[kt][c * 8 + 2] = (_Float16)u.z;  A1[kt][c * 8 + 3] = (_Float16)u.w;
        A1[kt][c * 8 + 4] = (_Float16)v.x;  A1[kt][c * 8 + 5] = (_Float16)v.y;
        A1[kt][c * 8 + 6] = (_Float16)v.z;  A1[kt][c * 8 + 7] = (_Float16)v.w;
      }
    }

#pragma unroll
    for (int mlp = 0; mlp < 2; ++mlp) {
      const _Float16* W  = sW + mlp * WSTRIDE;
      const float*    Bb = sB + mlp * BSTRIDE;
      float*          op = out + (size_t)mlp * E;

      // ---- layer 1: 128 -> 64 ----
#pragma unroll
      for (int nt = 0; nt < 4; ++nt) {
        float bv = Bb[nt * 16 + lm];
        v8f acc;
#pragma unroll
        for (int j = 0; j < 8; ++j) acc[j] = bv;
#pragma unroll
        for (int kt = 0; kt < 4; ++kt) {
          v16h bm = *(const v16h*)(W + OW1 + (nt * 16 + lm) * 128 + kt * 32 + half * 16);
          acc = wmma16(A1[kt], bm, acc);
        }
#pragma unroll
        for (int j = 0; j < 8; ++j)
          act[(half * 8 + j) * 64 + nt * 16 + lm] = (_Float16)lrelu(acc[j]);
      }

      // ---- layer 2: 64 -> 64 ----
      v16h A2[2];
#pragma unroll
      for (int kt = 0; kt < 2; ++kt)
#pragma unroll
        for (int c = 0; c < 2; ++c) {
          v8h q = *(const v8h*)(act + lm * 64 + kt * 32 + half * 16 + c * 8);
#pragma unroll
          for (int w = 0; w < 8; ++w) A2[kt][c * 8 + w] = q[w];
        }
#pragma unroll
      for (int nt = 0; nt < 4; ++nt) {
        float bv = Bb[64 + nt * 16 + lm];
        v8f acc;
#pragma unroll
        for (int j = 0; j < 8; ++j) acc[j] = bv;
#pragma unroll
        for (int kt = 0; kt < 2; ++kt) {
          v16h bm = *(const v16h*)(W + OW2 + (nt * 16 + lm) * 64 + kt * 32 + half * 16);
          acc = wmma16(A2[kt], bm, acc);
        }
#pragma unroll
        for (int j = 0; j < 8; ++j)
          act[(half * 8 + j) * 64 + nt * 16 + lm] = (_Float16)lrelu(acc[j]);
      }

      // ---- layer 3: 64 -> 32 ----
      v16h A3[2];
#pragma unroll
      for (int kt = 0; kt < 2; ++kt)
#pragma unroll
        for (int c = 0; c < 2; ++c) {
          v8h q = *(const v8h*)(act + lm * 64 + kt * 32 + half * 16 + c * 8);
#pragma unroll
          for (int w = 0; w < 8; ++w) A3[kt][c * 8 + w] = q[w];
        }
#pragma unroll
      for (int nt = 0; nt < 2; ++nt) {
        float bv = Bb[128 + nt * 16 + lm];
        v8f acc;
#pragma unroll
        for (int j = 0; j < 8; ++j) acc[j] = bv;
#pragma unroll
        for (int kt = 0; kt < 2; ++kt) {
          v16h bm = *(const v16h*)(W + OW3 + (nt * 16 + lm) * 64 + kt * 32 + half * 16);
          acc = wmma16(A3[kt], bm, acc);
        }
#pragma unroll
        for (int j = 0; j < 8; ++j)
          act[(half * 8 + j) * 64 + nt * 16 + lm] = (_Float16)lrelu(acc[j]);
      }

      // ---- layer 4: 32 -> 1 (padded 32x16 B, only column 0 valid) ----
      v16h A4;
#pragma unroll
      for (int c = 0; c < 2; ++c) {
        v8h q = *(const v8h*)(act + lm * 64 + half * 16 + c * 8);
#pragma unroll
        for (int w = 0; w < 8; ++w) A4[c * 8 + w] = q[w];
      }
      float b4 = (lm == 0) ? Bb[160] : 0.f;
      v8f acc4;
#pragma unroll
      for (int j = 0; j < 8; ++j) acc4[j] = b4;
      v16h bm4 = *(const v16h*)(W + OW4 + lm * 32 + half * 16);
      acc4 = wmma16(A4, bm4, acc4);

      if (lm == 0) {
#pragma unroll
        for (int j = 0; j < 8; ++j) {
          int ee = tileBase + half * 8 + j;
          if (ee < E) op[ee] = acc4[j];
        }
      }
    }
  }
}

extern "C" void kernel_launch(void* const* d_in, const int* in_sizes, int n_in,
                              void* d_out, int out_size, void* d_ws, size_t ws_size,
                              hipStream_t stream) {
  (void)n_in; (void)d_ws; (void)ws_size; (void)out_size;
  const float* x_src = (const float*)d_in[0];
  const float* x_dst = (const float*)d_in[1];
  const int*   eli   = (const int*)d_in[2];
  const int E = in_sizes[2] / 2;

  const float* ew1 = (const float*)d_in[3];  const float* eb1 = (const float*)d_in[4];
  const float* ww1 = (const float*)d_in[5];  const float* wb1 = (const float*)d_in[6];
  const float* ew2 = (const float*)d_in[7];  const float* eb2 = (const float*)d_in[8];
  const float* ww2 = (const float*)d_in[9];  const float* wb2 = (const float*)d_in[10];
  const float* ew3 = (const float*)d_in[11]; const float* eb3 = (const float*)d_in[12];
  const float* ww3 = (const float*)d_in[13]; const float* wb3 = (const float*)d_in[14];
  const float* ew4 = (const float*)d_in[15]; const float* eb4 = (const float*)d_in[16];
  const float* ww4 = (const float*)d_in[17]; const float* wb4 = (const float*)d_in[18];

  const int numTiles = (E + 15) >> 4;
  int blocks = 2048;
  const int maxBlocks = (numTiles + WAVES_PER_BLOCK - 1) / WAVES_PER_BLOCK;
  if (blocks > maxBlocks) blocks = maxBlocks;
  if (blocks < 1) blocks = 1;

  linkmlp_kernel<<<blocks, THREADS, 0, stream>>>(
      x_src, x_dst, eli,
      ew1, eb1, ww1, wb1, ew2, eb2, ww2, wb2,
      ew3, eb3, ww3, wb3, ew4, eb4, ww4, wb4,
      (float*)d_out, E);
}
